// SimpleSelfAttention_32856499814595
// MI455X (gfx1250) — compile-verified, matrix operand order checked by reading, not run on hardware
//
#include <hip/hip_runtime.h>
#include <hip/hip_bf16.h>

// ---------------------------------------------------------------------------
// LoRA self-attention for MI455X (gfx1250): bf16 WMMA + Tensor Data Mover.
// B=4, S=2048, H=1024, NH=16, HD=64, R=2, SCALING=2.0
// ---------------------------------------------------------------------------

#define B_  4
#define S_  2048
#define H_  1024
#define NH_ 16
#define HD_ 64
#define R_  2
#define SCALING_ 2.0f

typedef __attribute__((ext_vector_type(16))) __bf16 v16bf;
typedef __attribute__((ext_vector_type(8)))  float  v8f;

#if defined(__has_builtin)
#if __has_builtin(__builtin_amdgcn_tensor_load_to_lds)
#define USE_TDM 1
#endif
#endif
#ifndef USE_TDM
#define USE_TDM 0
#endif

union AB16 {
    v16bf v;
    uint4 q[2];
};

__device__ __forceinline__ v8f wmma_bf16(const AB16& a, const AB16& b, v8f c) {
    return __builtin_amdgcn_wmma_f32_16x16x32_bf16(
        /*neg_a=*/false, a.v, /*neg_b=*/false, b.v,
        /*c_mod=*/(short)0, c, /*reuse_a=*/false, /*reuse_b=*/false);
}

__device__ __forceinline__ unsigned short f2bf(float f) {
    unsigned int u = __float_as_uint(f);
    u += 0x7FFFu + ((u >> 16) & 1u);            // round-to-nearest-even
    return (unsigned short)(u >> 16);
}

// Low 32 bits of a generic pointer to __shared__ = LDS byte offset.
__device__ __forceinline__ unsigned lds_off(const void* p) {
    return (unsigned)(unsigned long long)(size_t)p;
}

#if USE_TDM
typedef __attribute__((ext_vector_type(4))) unsigned int tdm_v4u;
typedef __attribute__((ext_vector_type(8))) int          tdm_v8i;
typedef __attribute__((ext_vector_type(4))) int          tdm_v4i;

// 2D tile load, global -> LDS, bf16 elements, optional LDS row padding.
// tile_w in elements (row length), tile_h rows, row_stride in elements.
// pad codes per D#: interval 0=2,1=4,2=8,3=16,4=32,... DWORDs; amount code n = n+1 DWORDs.
__device__ __forceinline__ void tdm_load_2d(unsigned lds, const void* gptr,
                                            unsigned tile_w, unsigned tile_h,
                                            unsigned row_stride,
                                            unsigned pad_interval_code,
                                            unsigned pad_amount_code) {
    const unsigned TD0 = 1u << 20;   // generous tensor extents (no clipping)
    const unsigned TD1 = 1u << 20;
    unsigned long long ga = (unsigned long long)(size_t)gptr;
    tdm_v4u g0;
    g0[0] = 1u;                                                   // count=1
    g0[1] = lds;                                                  // lds_addr
    g0[2] = (unsigned)ga;                                         // global[31:0]
    g0[3] = (unsigned)((ga >> 32) & 0x01FFFFFFu) | (2u << 30);    // global hi | type=2
    tdm_v8i g1;
    g1[0] = (int)((1u << 16) |                 // data_size = 2 bytes
                  (1u << 20) |                 // pad_enable
                  (pad_interval_code << 22) |
                  (pad_amount_code << 25));
    g1[1] = (int)((TD0 & 0xFFFFu) << 16);                         // tensor_dim0 lo
    g1[2] = (int)((TD0 >> 16) | ((TD1 & 0xFFFFu) << 16));         // dim0 hi | dim1 lo
    g1[3] = (int)((TD1 >> 16) | (tile_w << 16));                  // dim1 hi | tile_dim0
    g1[4] = (int)(tile_h & 0xFFFFu);                              // tile_dim1 (tile_dim2=0)
    g1[5] = (int)row_stride;                                      // tensor_dim0_stride lo
    g1[6] = 0;
    g1[7] = 0;
    tdm_v4i gz4 = {0, 0, 0, 0};
    tdm_v8i gz8 = {0, 0, 0, 0, 0, 0, 0, 0};
    __builtin_amdgcn_tensor_load_to_lds(g0, g1, gz4, gz4, gz8, 0);
}
#endif

// ---------------------------------------------------------------------------
// fp32 -> bf16 elementwise convert (n divisible by 4)
// ---------------------------------------------------------------------------
__global__ __launch_bounds__(256) void cvt_bf16_k(const float* __restrict__ in,
                                                  unsigned short* __restrict__ out,
                                                  int n) {
    int i = (blockIdx.x * 256 + threadIdx.x) * 4;
    if (i < n) {
        float4 f = *(const float4*)&in[i];
        out[i + 0] = f2bf(f.x);
        out[i + 1] = f2bf(f.y);
        out[i + 2] = f2bf(f.z);
        out[i + 3] = f2bf(f.w);
    }
}

// ---------------------------------------------------------------------------
// W_eff = W + SCALING * (Bm @ A)  (rank-2), converted to bf16.
// ---------------------------------------------------------------------------
__global__ __launch_bounds__(256) void prep_w_k(const float* __restrict__ W,
                                                const float* __restrict__ A,
                                                const float* __restrict__ Bm,
                                                unsigned short* __restrict__ out,
                                                int hasLora) {
    int i = blockIdx.x * 256 + threadIdx.x;     // [0, H*H)
    int row = i >> 10;
    int col = i & (H_ - 1);
    float w = W[i];
    if (hasLora) {
        w += SCALING_ * (Bm[row * R_ + 0] * A[col] +
                         Bm[row * R_ + 1] * A[H_ + col]);
    }
    out[i] = f2bf(w);
}

// ---------------------------------------------------------------------------
// GEMM: Y[m][n] = sum_k Abf[m][k] * Wbf[n][k] + bias[n]
//   M = 8192, N = 1024, K = 1024, bf16 in, fp32 acc.
// Block 256 thr (8 waves), tile 128(M) x 64(N), wave tile 32x32.
// K staged 64 wide in LDS (TDM-loaded), 8 WMMAs per stage.
// ---------------------------------------------------------------------------
__global__ __launch_bounds__(256) void gemm_bf16_k(const unsigned short* __restrict__ Abf,
                                                   const unsigned short* __restrict__ Wbf,
                                                   const float* __restrict__ bias,
                                                   unsigned short* __restrict__ outHS,
                                                   float* __restrict__ outF) {
    __shared__ __align__(16) unsigned short Atile[128 * 72];  // 64 halfs + 8 pad
    __shared__ __align__(16) unsigned short Btile[64 * 72];

    const int tid   = threadIdx.x;
    const int lane  = tid & 31;
    const int wave  = tid >> 5;
    const int waveM = wave & 3;
    const int waveN = wave >> 2;
    const int Mbase = blockIdx.x * 128;
    const int Nbase = blockIdx.y * 64;
    const int hi    = lane >> 4;
    const int l15   = lane & 15;
    const int kb    = hi * 8;

    v8f acc00 = {}, acc01 = {}, acc10 = {}, acc11 = {};

    for (int k0 = 0; k0 < H_; k0 += 64) {
        __syncthreads();
#if USE_TDM
        if (wave == 0) {
            // 64-elem rows = 32 DWORDs -> pad_interval code 4; pad 4 DWORDs -> code 3
            tdm_load_2d(lds_off(Atile), &Abf[(size_t)(Mbase)*H_ + k0], 64, 128, H_, 4, 3);
            tdm_load_2d(lds_off(Btile), &Wbf[(size_t)(Nbase)*H_ + k0], 64, 64,  H_, 4, 3);
            __builtin_amdgcn_s_wait_tensorcnt(0);
        }
#else
#pragma unroll
        for (int i = 0; i < 4; ++i) {            // A: 128 rows x 8 chunks
            int c = tid + i * 256;
            int r = c >> 3, cc = c & 7;
            *(uint4*)&Atile[r * 72 + cc * 8] =
                *(const uint4*)&Abf[(Mbase + r) * H_ + k0 + cc * 8];
        }
#pragma unroll
        for (int i = 0; i < 2; ++i) {            // B: 64 rows x 8 chunks
            int c = tid + i * 256;
            int r = c >> 3, cc = c & 7;
            *(uint4*)&Btile[r * 72 + cc * 8] =
                *(const uint4*)&Wbf[(Nbase + r) * H_ + k0 + cc * 8];
        }
#endif
        __syncthreads();

#pragma unroll
        for (int ks = 0; ks < 2; ++ks) {
            const int ko = ks * 32 + kb;
            AB16 a0, a1, b0, b1;
            a0.q[0] = *(const uint4*)&Atile[(waveM * 32 + l15) * 72 + ko];
            a0.q[1] = *(const uint4*)&Atile[(waveM * 32 + l15) * 72 + ko + 16];
            a1.q[0] = *(const uint4*)&Atile[(waveM * 32 + 16 + l15) * 72 + ko];
            a1.q[1] = *(const uint4*)&Atile[(waveM * 32 + 16 + l15) * 72 + ko + 16];
            b0.q[0] = *(const uint4*)&Btile[(waveN * 32 + l15) * 72 + ko];
            b0.q[1] = *(const uint4*)&Btile[(waveN * 32 + l15) * 72 + ko + 16];
            b1.q[0] = *(const uint4*)&Btile[(waveN * 32 + 16 + l15) * 72 + ko];
            b1.q[1] = *(const uint4*)&Btile[(waveN * 32 + 16 + l15) * 72 + ko + 16];

            acc00 = wmma_bf16(a0, b0, acc00);
            acc01 = wmma_bf16(a0, b1, acc01);
            acc10 = wmma_bf16(a1, b0, acc10);
            acc11 = wmma_bf16(a1, b1, acc11);
        }
    }

    const float bias0 = bias[Nbase + waveN * 32 + l15];
    const float bias1 = bias[Nbase + waveN * 32 + 16 + l15];
    const int n0 = Nbase + waveN * 32 + l15;
    const int n1 = n0 + 16;

#pragma unroll
    for (int r = 0; r < 8; ++r) {
        int m0 = Mbase + waveM * 32 + r + 8 * hi;
        int m1 = m0 + 16;
        float v00 = acc00[r] + bias0;
        float v01 = acc01[r] + bias1;
        float v10 = acc10[r] + bias0;
        float v11 = acc11[r] + bias1;
        if (outF) {
            outF[(size_t)m0 * H_ + n0] = v00;
            outF[(size_t)m0 * H_ + n1] = v01;
            outF[(size_t)m1 * H_ + n0] = v10;
            outF[(size_t)m1 * H_ + n1] = v11;
        } else {
            int b0i = m0 >> 11, s0i = m0 & (S_ - 1);
            int b1i = m1 >> 11, s1i = m1 & (S_ - 1);
            int h0 = n0 >> 6, d0 = n0 & 63;
            int h1 = n1 >> 6, d1 = n1 & 63;
            outHS[(((size_t)b0i * NH_ + h0) * S_ + s0i) * HD_ + d0] = f2bf(v00);
            outHS[(((size_t)b0i * NH_ + h1) * S_ + s0i) * HD_ + d1] = f2bf(v01);
            outHS[(((size_t)b1i * NH_ + h0) * S_ + s1i) * HD_ + d0] = f2bf(v10);
            outHS[(((size_t)b1i * NH_ + h1) * S_ + s1i) * HD_ + d1] = f2bf(v11);
        }
    }
}

// ---------------------------------------------------------------------------
// Flash attention: one block = 64 query rows of one (b,h); 4 waves x 16 rows.
// K tile (32 keys x 64d) TDM-loaded; V transposed manually; online softmax.
// ---------------------------------------------------------------------------
__global__ __launch_bounds__(128) void attn_k(const unsigned short* __restrict__ Qb,
                                              const unsigned short* __restrict__ Kb,
                                              const unsigned short* __restrict__ Vb,
                                              const int* __restrict__ mask,
                                              unsigned short* __restrict__ ctxb) {
    __shared__ __align__(16) unsigned short Kt[32 * 72];        // [key][d], +8 pad
    __shared__ __align__(16) unsigned short Vt[64 * 40];        // [d][key], +8 pad
    __shared__ __align__(16) unsigned short Pbuf[4 * 16 * 32];  // per-wave P

    const int tid  = threadIdx.x;
    const int lane = tid & 31;
    const int wave = tid >> 5;
    const int h    = blockIdx.y;
    const int b    = blockIdx.z;
    const int q0   = blockIdx.x * 64 + wave * 16;
    const int hi   = lane >> 4;
    const int l15  = lane & 15;
    const int kbs  = hi * 8;

    const size_t headOff = ((size_t)b * NH_ + h) * S_ * HD_;
    const unsigned short* Qh = Qb + headOff;
    const unsigned short* Kh = Kb + headOff;
    const unsigned short* Vh = Vb + headOff;

    AB16 qf0, qf1;
    {
        const unsigned short* qrow = Qh + (size_t)(q0 + l15) * HD_;
        qf0.q[0] = *(const uint4*)&qrow[kbs];
        qf0.q[1] = *(const uint4*)&qrow[kbs + 16];
        qf1.q[0] = *(const uint4*)&qrow[32 + kbs];
        qf1.q[1] = *(const uint4*)&qrow[32 + kbs + 16];
    }

    float mrow[8], lrow[8];
#pragma unroll
    for (int r = 0; r < 8; ++r) { mrow[r] = -1e30f; lrow[r] = 0.f; }
    v8f o0 = {}, o1 = {}, o2 = {}, o3 = {};

    for (int kb0 = 0; kb0 < S_; kb0 += 32) {
        __syncthreads();
#if USE_TDM
        if (wave == 0) {
            // K tile rows of 64 halfs (32 DWORDs) -> interval code 4, pad code 3
            tdm_load_2d(lds_off(Kt), &Kh[(size_t)kb0 * HD_], 64, 32, HD_, 4, 3);
        }
#else
#pragma unroll
        for (int i = 0; i < 2; ++i) {
            int c = tid + i * 128;
            int r = c >> 3, cc = c & 7;
            *(uint4*)&Kt[r * 72 + cc * 8] =
                *(const uint4*)&Kh[(size_t)(kb0 + r) * HD_ + cc * 8];
        }
#endif
        // V tile transposed into Vt[d][key] (all threads)
#pragma unroll
        for (int i = 0; i < 2; ++i) {
            int c = tid + i * 128;
            int r = c >> 3, cc = c & 7;
            uint4 raw = *(const uint4*)&Vh[(size_t)(kb0 + r) * HD_ + cc * 8];
            const unsigned short* hs = (const unsigned short*)&raw;
#pragma unroll
            for (int j = 0; j < 8; ++j) Vt[(cc * 8 + j) * 40 + r] = hs[j];
        }
#if USE_TDM
        if (wave == 0) __builtin_amdgcn_s_wait_tensorcnt(0);
#endif
        __syncthreads();

        const float mb0 = mask[b * S_ + kb0 + l15]      ? 0.f : -1e9f;
        const float mb1 = mask[b * S_ + kb0 + 16 + l15] ? 0.f : -1e9f;

        // scores S = Q @ K^T : two 16x16 tiles, K-dim 64
        AB16 kf;
        v8f s0 = {}, s1 = {};
        kf.q[0] = *(const uint4*)&Kt[l15 * 72 + kbs];
        kf.q[1] = *(const uint4*)&Kt[l15 * 72 + kbs + 16];
        s0 = wmma_bf16(qf0, kf, s0);
        kf.q[0] = *(const uint4*)&Kt[l15 * 72 + 32 + kbs];
        kf.q[1] = *(const uint4*)&Kt[l15 * 72 + 32 + kbs + 16];
        s0 = wmma_bf16(qf1, kf, s0);
        kf.q[0] = *(const uint4*)&Kt[(16 + l15) * 72 + kbs];
        kf.q[1] = *(const uint4*)&Kt[(16 + l15) * 72 + kbs + 16];
        s1 = wmma_bf16(qf0, kf, s1);
        kf.q[0] = *(const uint4*)&Kt[(16 + l15) * 72 + 32 + kbs];
        kf.q[1] = *(const uint4*)&Kt[(16 + l15) * 72 + 32 + kbs + 16];
        s1 = wmma_bf16(qf1, kf, s1);

        unsigned short* Pw = &Pbuf[wave * 16 * 32];
#pragma unroll
        for (int r = 0; r < 8; ++r) {
            float x0 = s0[r] * 0.125f + mb0;   // 1/sqrt(64)
            float x1 = s1[r] * 0.125f + mb1;
            float t = fmaxf(x0, x1);
            t = fmaxf(t, __shfl_xor(t, 1, 32));
            t = fmaxf(t, __shfl_xor(t, 2, 32));
            t = fmaxf(t, __shfl_xor(t, 4, 32));
            t = fmaxf(t, __shfl_xor(t, 8, 32));
            float mnew  = fmaxf(mrow[r], t);
            float alpha = __expf(mrow[r] - mnew);
            float p0 = __expf(x0 - mnew);
            float p1 = __expf(x1 - mnew);
            float ps = p0 + p1;
            ps += __shfl_xor(ps, 1, 32);
            ps += __shfl_xor(ps, 2, 32);
            ps += __shfl_xor(ps, 4, 32);
            ps += __shfl_xor(ps, 8, 32);
            lrow[r] = lrow[r] * alpha + ps;
            mrow[r] = mnew;
            o0[r] *= alpha; o1[r] *= alpha; o2[r] *= alpha; o3[r] *= alpha;
            int row = r + 8 * hi;
            Pw[row * 32 + l15]      = f2bf(p0);
            Pw[row * 32 + 16 + l15] = f2bf(p1);
        }

        // reload P as 16x32 A-fragment, then O += P @ V
        AB16 pf, vf;
        pf.q[0] = *(const uint4*)&Pw[l15 * 32 + kbs];
        pf.q[1] = *(const uint4*)&Pw[l15 * 32 + kbs + 16];
        vf.q[0] = *(const uint4*)&Vt[(0 + l15) * 40 + kbs];
        vf.q[1] = *(const uint4*)&Vt[(0 + l15) * 40 + kbs + 16];
        o0 = wmma_bf16(pf, vf, o0);
        vf.q[0] = *(const uint4*)&Vt[(16 + l15) * 40 + kbs];
        vf.q[1] = *(const uint4*)&Vt[(16 + l15) * 40 + kbs + 16];
        o1 = wmma_bf16(pf, vf, o1);
        vf.q[0] = *(const uint4*)&Vt[(32 + l15) * 40 + kbs];
        vf.q[1] = *(const uint4*)&Vt[(32 + l15) * 40 + kbs + 16];
        o2 = wmma_bf16(pf, vf, o2);
        vf.q[0] = *(const uint4*)&Vt[(48 + l15) * 40 + kbs];
        vf.q[1] = *(const uint4*)&Vt[(48 + l15) * 40 + kbs + 16];
        o3 = wmma_bf16(pf, vf, o3);
    }

#pragma unroll
    for (int r = 0; r < 8; ++r) {
        float inv = (lrow[r] > 0.f) ? (1.0f / lrow[r]) : 0.f;
        int q = q0 + r + 8 * hi;
        size_t base = ((size_t)b * S_ + q) * H_ + (size_t)h * HD_;
        ctxb[base + 0  + l15] = f2bf(o0[r] * inv);
        ctxb[base + 16 + l15] = f2bf(o1[r] * inv);
        ctxb[base + 32 + l15] = f2bf(o2[r] * inv);
        ctxb[base + 48 + l15] = f2bf(o3[r] * inv);
    }
}

// ---------------------------------------------------------------------------
extern "C" void kernel_launch(void* const* d_in, const int* in_sizes, int n_in,
                              void* d_out, int out_size, void* d_ws, size_t ws_size,
                              hipStream_t stream) {
    const float* x    = (const float*)d_in[0];
    const int*   mask = (const int*)  d_in[1];
    const float* Wq   = (const float*)d_in[2];
    const float* bq   = (const float*)d_in[3];
    const float* Wk   = (const float*)d_in[4];
    const float* bk   = (const float*)d_in[5];
    const float* Wv   = (const float*)d_in[6];
    const float* bv   = (const float*)d_in[7];
    const float* Wo   = (const float*)d_in[8];
    const float* bo   = (const float*)d_in[9];
    const float* Aq   = (const float*)d_in[10];
    const float* Bq   = (const float*)d_in[11];
    const float* Ak   = (const float*)d_in[12];
    const float* Bk   = (const float*)d_in[13];
    const float* Av   = (const float*)d_in[14];
    const float* Bv   = (const float*)d_in[15];
    float* out = (float*)d_out;

    const size_t nTok = (size_t)B_ * S_ * H_;      // 8,388,608
    const size_t nW   = (size_t)H_ * H_;           // 1,048,576

    char* ws = (char*)d_ws;
    unsigned short* xb   = (unsigned short*)ws; ws += nTok * 2;
    unsigned short* wqE  = (unsigned short*)ws; ws += nW * 2;
    unsigned short* wkE  = (unsigned short*)ws; ws += nW * 2;
    unsigned short* wvE  = (unsigned short*)ws; ws += nW * 2;
    unsigned short* woE  = (unsigned short*)ws; ws += nW * 2;
    unsigned short* qb   = (unsigned short*)ws; ws += nTok * 2;
    unsigned short* kbuf = (unsigned short*)ws; ws += nTok * 2;
    unsigned short* vbuf = (unsigned short*)ws; ws += nTok * 2;
    unsigned short* ctxb = (unsigned short*)ws; ws += nTok * 2;

    // 1) precision prep
    cvt_bf16_k<<<(int)(nTok / 1024), 256, 0, stream>>>(x, xb, (int)nTok);
    prep_w_k<<<(int)(nW / 256), 256, 0, stream>>>(Wq, Aq, Bq, wqE, 1);
    prep_w_k<<<(int)(nW / 256), 256, 0, stream>>>(Wk, Ak, Bk, wkE, 1);
    prep_w_k<<<(int)(nW / 256), 256, 0, stream>>>(Wv, Av, Bv, wvE, 1);
    prep_w_k<<<(int)(nW / 256), 256, 0, stream>>>(Wo, Aq, Bq, woE, 0);

    // 2) Q/K/V projections (bf16 head-split outputs)
    dim3 ggrid(64, 16);   // M/128 x N/64
    gemm_bf16_k<<<ggrid, 256, 0, stream>>>(xb, wqE, bq, qb,   nullptr);
    gemm_bf16_k<<<ggrid, 256, 0, stream>>>(xb, wkE, bk, kbuf, nullptr);
    gemm_bf16_k<<<ggrid, 256, 0, stream>>>(xb, wvE, bv, vbuf, nullptr);

    // 3) flash attention
    dim3 agrid(S_ / 64, NH_, B_);
    attn_k<<<agrid, 128, 0, stream>>>(qb, kbuf, vbuf, mask, ctxb);

    // 4) output projection (fp32 flat output)
    gemm_bf16_k<<<ggrid, 256, 0, stream>>>(ctxb, woE, bo, nullptr, out);
}
